// FeatureAggIBRNet_38646115729533
// MI455X (gfx1250) — matrix operand order repaired and checked by reading, not compiled, not run on hardware
//
#include <hip/hip_runtime.h>
#include <hip/hip_bf16.h>

typedef _Float16 v16h __attribute__((ext_vector_type(16)));
typedef _Float16 v8h  __attribute__((ext_vector_type(8)));
typedef float    v8f  __attribute__((ext_vector_type(8)));
typedef int      vi4  __attribute__((ext_vector_type(4)));

// Fragment offsets in d_ws (units of halves, 512 halves per 16x16x32 B-fragment)
#define OFF_VIEW1 (512*0)
#define OFF_VIEW2 (512*1)
#define OFF_BASE1 (512*3)
#define OFF_BASE2 (512*15)
#define OFF_VIS1A (512*19)
#define OFF_VIS1B (512*21)
#define OFF_VIS2A (512*24)
#define OFF_VIS2B (512*26)
#define OFF_GEO1  (512*27)
#define OFF_GEO2  (512*39)
#define OFF_OUT1  (512*41)
#define OFF_OUT2  (512*42)

#define RPB 8      // rays per workgroup
#define NW  4      // waves per workgroup
#define NS  8      // samples (views) per ray
#define XROW (RPB*37 + 8)   // padded per-view row in xbuf (floats), 16B-aligned stride

// xbuf is [view][ray*37 + c] so each view's 8*37 floats are contiguous for async B128 copies
#define XB(wr, sm, c) xbuf[sm][(wr)*37 + (c)]

#define HAS_ASYNC_LD  __has_builtin(__builtin_amdgcn_global_load_async_to_lds_b128)
#define HAS_ASYNC_ST  __has_builtin(__builtin_amdgcn_global_store_async_from_lds_b128)

// async builtins take int4 pointers: global (AS1) and LDS (AS3)
typedef __attribute__((address_space(1))) vi4 GV4;
typedef __attribute__((address_space(3))) vi4 LV4;

__device__ __forceinline__ void wait_async0() {
#if __has_builtin(__builtin_amdgcn_s_wait_asynccnt)
    __builtin_amdgcn_s_wait_asynccnt(0);
#else
    asm volatile("s_wait_asynccnt 0" ::: "memory");
#endif
}

__device__ __forceinline__ float elu1(float x) { return x > 0.f ? x : (__expf(x) - 1.f); }

// Build an A fragment (16x32 f16) from a row-major f16 LDS activation buffer.
// ISA layout: lane l holds row M=l%16; element e -> K = kt*32 + e + 8*(e>=8) + 8*(l>=16)
// => two contiguous runs of 8 halves at K = k0.. and K = k0+16.. with k0 = 8*(l>=16).
__device__ __forceinline__ v16h load_a_frag(const _Float16* base, int stride, int kt, int lane) {
    const int row = lane & 15;
    const int k0  = kt * 32 + ((lane & 16) ? 8 : 0);
    const _Float16* p = base + row * stride + k0;
    v8h lo = *(const v8h*)(p);
    v8h hi = *(const v8h*)(p + 16);
    v16h a;
#pragma unroll
    for (int i = 0; i < 8; ++i) { a[i] = lo[i]; a[8 + i] = hi[i]; }
    return a;
}

// Generic WMMA MLP layer: act(A[16xK] * W[KxN] + b) -> f16 LDS, f32 accumulation.
// ACT: 0 = none, 1 = elu, 2 = relu
template<int KT, int NT, int ACT>
__device__ __forceinline__ void mlp_layer(const _Float16* aIn, int sIn,
                                          _Float16* aOut, int sOut, int colOff,
                                          const _Float16* __restrict__ wfrag,
                                          const float* __restrict__ bias,
                                          int Nreal, int lane)
{
    v16h A[KT];
#pragma unroll
    for (int kt = 0; kt < KT; ++kt) A[kt] = load_a_frag(aIn, sIn, kt, lane);
    const int n  = lane & 15;
    const int rb = (lane & 16) ? 8 : 0;
#pragma unroll
    for (int nt = 0; nt < NT; ++nt) {
        const int ng = nt * 16 + n;
        const float bv = (ng < Nreal) ? bias[ng] : 0.f;
        v8f c;
#pragma unroll
        for (int j = 0; j < 8; ++j) c[j] = bv;
#pragma unroll
        for (int kt = 0; kt < KT; ++kt) {
            v16h B = *(const v16h*)(wfrag + (size_t)(nt * KT + kt) * 512 + lane * 16);
            c = __builtin_amdgcn_wmma_f32_16x16x32_f16(false, A[kt], false, B,
                                                       (short)0, c, false, false);
        }
#pragma unroll
        for (int j = 0; j < 8; ++j) {
            float v = c[j];
            if (ACT == 1) v = elu1(v);
            if (ACT == 2) v = fmaxf(v, 0.f);
            aOut[(rb + j) * sOut + colOff + ng] = (_Float16)v;
        }
    }
}

// ---------------- setup: swizzle fp32 weights into f16 B-fragment layout ----------------
__global__ void fa_setup(const float* vw1, const float* vw2, const float* bw1, const float* bw2,
                         const float* c1w1, const float* c1w2, const float* c2w1, const float* c2w2,
                         const float* gw1, const float* gw2, const float* ow1, const float* ow2,
                         _Float16* __restrict__ wf)
{
    struct LD { const float* W; int K, N, KT, NT, off; };
    const LD t[12] = {
        { vw1,  4, 16, 1, 1, OFF_VIEW1 },
        { vw2, 16, 32, 1, 2, OFF_VIEW2 },
        { bw1, 96, 64, 3, 4, OFF_BASE1 },
        { bw2, 64, 32, 2, 2, OFF_BASE2 },
        { c1w1,32, 32, 1, 2, OFF_VIS1A },
        { c1w2,32, 33, 1, 3, OFF_VIS1B },
        { c2w1,32, 32, 1, 2, OFF_VIS2A },
        { c2w2,32,  1, 1, 1, OFF_VIS2B },
        { gw1, 65, 64, 3, 4, OFF_GEO1  },
        { gw2, 64, 16, 2, 1, OFF_GEO2  },
        { ow1, 16, 16, 1, 1, OFF_OUT1  },
        { ow2, 16,  1, 1, 1, OFF_OUT2  },
    };
    const int tid  = blockIdx.x * blockDim.x + threadIdx.x;
    const int nthr = gridDim.x * blockDim.x;
    for (int L = 0; L < 12; ++L) {
        const LD d = t[L];
        const int total = d.KT * d.NT * 512;
        for (int i = tid; i < total; i += nthr) {
            const int frag = i >> 9, r = i & 511;
            const int lanei = r >> 4, e = r & 15;
            const int kt = frag % d.KT, nt = frag / d.KT;
            // B layout: lane l, element e -> K = kt*32 + (e&7) + 16*(e>=8) + 8*(l>=16), N = nt*16 + (l%16)
            const int K = kt * 32 + (e & 7) + ((e >= 8) ? 16 : 0) + ((lanei >= 16) ? 8 : 0);
            const int N = nt * 16 + (lanei & 15);
            const float v = (K < d.K && N < d.N) ? d.W[K * d.N + N] : 0.f;
            wf[d.off + i] = (_Float16)v;
        }
    }
}

// ---------------- fused main kernel: 8 rays per WG, 2 rays per wave ----------------
__global__ __launch_bounds__(128)
void fa_main(const float* __restrict__ x, const float* __restrict__ sp,
             const _Float16* __restrict__ wf,
             const float* vb1, const float* vb2, const float* bb1, const float* bb2,
             const float* c1b1, const float* c1b2, const float* c2b1, const float* c2b2,
             const float* gb1, const float* gb2, const float* ob1, const float* ob2,
             float* __restrict__ out, int R)
{
    __shared__ __align__(16) float    xbuf[NS][XROW];      // [view][ray*37+c]
    __shared__ __align__(16) _Float16 P0[NW][16][104];     // ping
    __shared__ __align__(16) _Float16 P1[NW][16][104];     // pong
    __shared__ __align__(16) _Float16 IFb[NW][16][40];     // persistent img_feat
    __shared__ __align__(16) _Float16 geoH[16][96];        // geo head input (rows=WG rays)
    __shared__ float wrow[NW][16], visrow[NW][16], vwrow[NW][16], grow[NW][16];
    __shared__ float masksum[RPB];

    const int tid  = threadIdx.x;
    const int lane = tid & 31;
    const int wid  = tid >> 5;
    const int ray0 = blockIdx.x * RPB;
    const float sAbs = fabsf(sp[0]);

    // stream x in: per view, 8 rays * 37 floats = 1184B contiguous & 16B aligned
#if HAS_ASYNC_LD
    for (int v = 0; v < NS; ++v) {
        const float* src = x + ((size_t)v * R + ray0) * 37;
        __builtin_prefetch(src + RPB * 37, 0, 0);          // next block's segment
        for (int i = tid; i < (RPB * 37) / 4; i += 128)    // 74 x B128
            __builtin_amdgcn_global_load_async_to_lds_b128(
                (GV4*)(src + i * 4), (LV4*)&xbuf[v][i * 4], 0, 0);
    }
#else
    for (int v = 0; v < NS; ++v) {
        const float* src = x + ((size_t)v * R + ray0) * 37;
        for (int i = tid; i < RPB * 37; i += 128) xbuf[v][i] = src[i];
    }
#endif

    for (int i = tid; i < 16 * 96; i += 128) (&geoH[0][0])[i] = (_Float16)0.f;

#if HAS_ASYNC_LD
    wait_async0();
#endif
    __syncthreads();

    // ---- phase 1: per-row blend weights (softmax-like over 8 samples) ----
    {
        const int m  = lane & 15;
        const int wr = 2 * wid + (m >> 3);
        const int sm = m & 7;
        const float mask = XB(wr, sm, 0);
        const float dot  = XB(wr, sm, 36);
        float ed = __expf(sAbs * (dot - 1.f));
        float mn = ed;
        mn = fminf(mn, __shfl_xor(mn, 1, 32));
        mn = fminf(mn, __shfl_xor(mn, 2, 32));
        mn = fminf(mn, __shfl_xor(mn, 4, 32));
        float wu = (ed - mn) * mask;
        float su = wu;
        su += __shfl_xor(su, 1, 32); su += __shfl_xor(su, 2, 32); su += __shfl_xor(su, 4, 32);
        const float w = wu / (su + 1e-8f);
        float ms = mask;
        ms += __shfl_xor(ms, 1, 32); ms += __shfl_xor(ms, 2, 32); ms += __shfl_xor(ms, 4, 32);
        if (lane < 16) {
            wrow[wid][m] = w;
            if (sm == 0) masksum[wr] = ms;
        }
    }
    // ---- phase 2: dirs -> P0 cols 0..31 (K padded with zeros) ----
    for (int i = lane; i < 16 * 32; i += 32) {
        const int m = i >> 5, c = i & 31;
        const int wr = 2 * wid + (m >> 3), sm = m & 7;
        P0[wid][m][c] = (_Float16)((c < 4) ? XB(wr, sm, 33 + c) : 0.f);
    }
    __syncthreads();

    // view MLP: 4->16 (elu), 16->32 (elu) then + img feat
    mlp_layer<1, 1, 1>(&P0[wid][0][0], 104, &P1[wid][0][0], 104, 0, wf + OFF_VIEW1, vb1, 16, lane);
    for (int i = lane; i < 16 * 16; i += 32) P1[wid][i >> 4][16 + (i & 15)] = (_Float16)0.f;
    __syncthreads();
    mlp_layer<1, 2, 1>(&P1[wid][0][0], 104, &P0[wid][0][0], 104, 64, wf + OFF_VIEW2, vb2, 32, lane);
    __syncthreads();
    for (int i = lane; i < 16 * 32; i += 32) {     // img_feat_rgb = feat + view_feat
        const int m = i >> 5, c = i & 31;
        const int wr = 2 * wid + (m >> 3), sm = m & 7;
        P0[wid][m][64 + c] = (_Float16)((float)P0[wid][m][64 + c] + XB(wr, sm, 1 + c));
    }
    __syncthreads();

    // ---- phase 3: weighted mean/var over samples -> P0 cols 0..63 ----
    {
        const int rayL = lane >> 4, m0 = rayL * 8;
#pragma unroll
        for (int h = 0; h < 2; ++h) {
            const int c = (lane & 15) + 16 * h;
            float mean = 0.f;
#pragma unroll
            for (int q = 0; q < 8; ++q) mean += wrow[wid][m0 + q] * (float)P0[wid][m0 + q][64 + c];
            float var = 0.f;
#pragma unroll
            for (int q = 0; q < 8; ++q) {
                const float d = (float)P0[wid][m0 + q][64 + c] - mean;
                var += wrow[wid][m0 + q] * d * d;
            }
#pragma unroll
            for (int q = 0; q < 8; ++q) {
                P0[wid][m0 + q][c]      = (_Float16)mean;
                P0[wid][m0 + q][32 + c] = (_Float16)var;
            }
        }
    }
    __syncthreads();

    // base MLP: 96->64 (elu), 64->32 (elu) -> IFb
    mlp_layer<3, 4, 1>(&P0[wid][0][0], 104, &P1[wid][0][0], 104, 0, wf + OFF_BASE1, bb1, 64, lane);
    __syncthreads();
    mlp_layer<2, 2, 1>(&P1[wid][0][0], 104, &IFb[wid][0][0], 40, 0, wf + OFF_BASE2, bb2, 32, lane);
    __syncthreads();
    for (int i = lane; i < 16 * 32; i += 32) {     // vis1 input = img_feat * weight
        const int m = i >> 5, c = i & 31;
        P0[wid][m][c] = (_Float16)((float)IFb[wid][m][c] * wrow[wid][m]);
    }
    __syncthreads();

    // vis1 MLP: 32->32 (elu), 32->33 (elu)
    mlp_layer<1, 2, 1>(&P0[wid][0][0], 104, &P1[wid][0][0], 104, 0, wf + OFF_VIS1A, c1b1, 32, lane);
    __syncthreads();
    mlp_layer<1, 3, 1>(&P1[wid][0][0], 104, &P0[wid][0][0], 104, 0, wf + OFF_VIS1B, c1b2, 33, lane);
    __syncthreads();
    if (lane < 16) {                               // gate = sigmoid(vis)*mask
        const int m = lane, wr = 2 * wid + (m >> 3), sm = m & 7;
        const float vis = (float)P0[wid][m][32];
        grow[wid][m] = (1.f / (1.f + __expf(-vis))) * XB(wr, sm, 0);
    }
    __syncthreads();
    for (int i = lane; i < 16 * 32; i += 32) {     // img_feat += res; vis2 input = img_feat*gate
        const int m = i >> 5, c = i & 31;
        const float nif = (float)IFb[wid][m][c] + (float)P0[wid][m][c];
        IFb[wid][m][c] = (_Float16)nif;
        P0[wid][m][c]  = (_Float16)(nif * grow[wid][m]);
    }
    __syncthreads();

    // vis2 MLP: 32->32 (elu), 32->1 (elu) * mask
    mlp_layer<1, 2, 1>(&P0[wid][0][0], 104, &P1[wid][0][0], 104, 0, wf + OFF_VIS2A, c2b1, 32, lane);
    __syncthreads();
    mlp_layer<1, 1, 1>(&P1[wid][0][0], 104, &P0[wid][0][0], 104, 0, wf + OFF_VIS2B, c2b2, 1, lane);
    __syncthreads();
    if (lane < 16) {
        const int m = lane, wr = 2 * wid + (m >> 3), sm = m & 7;
        visrow[wid][m] = (float)P0[wid][m][0] * XB(wr, sm, 0);
    }
    __syncthreads();

    // ---- phase 7: vis_weight, per-ray stats -> geoH; app rows -> xbuf ----
    {
        const int m = lane & 15;
        const float v = visrow[wid][m];
        float sv = v;
        sv += __shfl_xor(sv, 1, 32); sv += __shfl_xor(sv, 2, 32); sv += __shfl_xor(sv, 4, 32);
        const float vw = v / (sv + 1e-8f);
        float mv = vw;
        mv += __shfl_xor(mv, 1, 32); mv += __shfl_xor(mv, 2, 32); mv += __shfl_xor(mv, 4, 32);
        if (lane < 16) {
            vwrow[wid][m] = vw;
            if ((m & 7) == 0) geoH[2 * wid + (m >> 3)][64] = (_Float16)(mv * 0.125f);
        }
    }
    __syncthreads();
    {
        const int rayL = lane >> 4, m0 = rayL * 8, gr = 2 * wid + rayL;
#pragma unroll
        for (int h = 0; h < 2; ++h) {
            const int c = (lane & 15) + 16 * h;
            float mean = 0.f;
#pragma unroll
            for (int q = 0; q < 8; ++q) mean += vwrow[wid][m0 + q] * (float)IFb[wid][m0 + q][c];
            float var = 0.f;
#pragma unroll
            for (int q = 0; q < 8; ++q) {
                const float d = (float)IFb[wid][m0 + q][c] - mean;
                var += vwrow[wid][m0 + q] * d * d;
            }
            geoH[gr][c]      = (_Float16)mean;
            geoH[gr][32 + c] = (_Float16)var;
        }
    }
    for (int i = lane; i < 16 * 33; i += 32) {     // app row = [img_feat(32), vis, dirs(kept)]
        const int m = i / 33, c = i % 33;
        const int wr = 2 * wid + (m >> 3), sm = m & 7;
        XB(wr, sm, c) = (c < 32) ? (float)IFb[wid][m][c] : visrow[wid][m];
    }
    __syncthreads();

    // ---- geo/out head: one 16-row WMMA pass by wave 0 (rows 8..15 are zero) ----
    if (wid == 0) {
        mlp_layer<3, 4, 1>(&geoH[0][0], 96, &P1[0][0][0], 104, 0, wf + OFF_GEO1, gb1, 64, lane);
        __builtin_amdgcn_wave_barrier();
        mlp_layer<2, 1, 1>(&P1[0][0][0], 104, &P0[0][0][0], 104, 0, wf + OFF_GEO2, gb2, 16, lane);
        for (int i = lane; i < 256; i += 32) P0[0][i >> 4][16 + (i & 15)] = (_Float16)0.f;
        __builtin_amdgcn_wave_barrier();
        mlp_layer<1, 1, 1>(&P0[0][0][0], 104, &P1[0][0][0], 104, 0, wf + OFF_OUT1, ob1, 16, lane);
        for (int i = lane; i < 256; i += 32) P1[0][i >> 4][16 + (i & 15)] = (_Float16)0.f;
        __builtin_amdgcn_wave_barrier();
        mlp_layer<1, 1, 2>(&P1[0][0][0], 104, &P0[0][0][0], 104, 0, wf + OFF_OUT2, ob2, 1, lane);
        __builtin_amdgcn_wave_barrier();
        if (lane < 8) {
            float d = (float)P0[0][lane][0];
            if (masksum[lane] < 1.f) d = 0.f;
            out[ray0 + lane] = d;
        }
    }
    __syncthreads();

    // ---- app_feat store (layout [view][ray][37], after density block) ----
    float* outApp = out + R;
#if HAS_ASYNC_ST
    for (int v = 0; v < NS; ++v) {
        float* dst = outApp + ((size_t)v * R + ray0) * 37;
        for (int i = tid; i < (RPB * 37) / 4; i += 128)    // 74 x B128
            __builtin_amdgcn_global_store_async_from_lds_b128(
                (GV4*)(dst + i * 4), (LV4*)&xbuf[v][i * 4], 0, 0);
    }
    wait_async0();
#else
    for (int v = 0; v < NS; ++v) {
        float* dst = outApp + ((size_t)v * R + ray0) * 37;
        for (int i = tid; i < RPB * 37; i += 128) dst[i] = xbuf[v][i];
    }
#endif
}

extern "C" void kernel_launch(void* const* d_in, const int* in_sizes, int n_in,
                              void* d_out, int out_size, void* d_ws, size_t ws_size,
                              hipStream_t stream) {
    const float* x  = (const float*)d_in[0];
    const float* sp = (const float*)d_in[1];
    _Float16* wf = (_Float16*)d_ws;

    fa_setup<<<64, 256, 0, stream>>>(
        (const float*)d_in[2],  (const float*)d_in[4],
        (const float*)d_in[6],  (const float*)d_in[8],
        (const float*)d_in[10], (const float*)d_in[12],
        (const float*)d_in[14], (const float*)d_in[16],
        (const float*)d_in[18], (const float*)d_in[20],
        (const float*)d_in[22], (const float*)d_in[24], wf);

    const int R = in_sizes[0] / (8 * 37);
    fa_main<<<R / RPB, 128, 0, stream>>>(
        x, sp, wf,
        (const float*)d_in[3],  (const float*)d_in[5],
        (const float*)d_in[7],  (const float*)d_in[9],
        (const float*)d_in[11], (const float*)d_in[13],
        (const float*)d_in[15], (const float*)d_in[17],
        (const float*)d_in[19], (const float*)d_in[21],
        (const float*)d_in[23], (const float*)d_in[25],
        (float*)d_out, R);
}
